// DescriptorNetwork_6141803233554
// MI455X (gfx1250) — matrix operand order repaired and checked by reading, not compiled, not run on hardware
//
#include <hip/hip_runtime.h>

typedef __attribute__((ext_vector_type(16))) __bf16 v16bf;
typedef __attribute__((ext_vector_type(8)))  __bf16 v8bf;
typedef __attribute__((ext_vector_type(8)))  float  v8f;

namespace {

constexpr int kFea = 64;
constexpr int kC   = 25600;
constexpr int kNpc = 5;
constexpr int kN   = kC * kNpc;      // 128000 nodes
constexpr int kM   = kN * kNpc;      // 640000 edges
constexpr int kHid = 256;
constexpr int kEmb = 200;
constexpr float kSlope = 0.01f;      // jax.nn.leaky_relu default

__device__ __forceinline__ __bf16 tobf(float x) { return (__bf16)x; }

// ---------------------------------------------------------------------------
// Fragment helpers (CDNA5 WMMA 16x16x32 bf16, wave32; ISA 05_wmma layouts)
// A (16x32, 16-bit): lane<16 row=lane holds K {0..7,16..23}; lane>=16 row=lane-16
// holds K {8..15,24..31}.  B pre-swizzled so a fragment is 16 contiguous bf16
// per lane.  C/D: vgpr r, lane l -> row = r + 8*(l/16), col = l%16.
// ---------------------------------------------------------------------------
__device__ __forceinline__ v16bf load_a_frag(const __bf16* __restrict__ rowbase,
                                             int kstep, int lane) {
  int k0 = kstep * 32 + ((lane & 16) ? 8 : 0);
  v8bf lo = *(const v8bf*)(rowbase + k0);
  v8bf hi = *(const v8bf*)(rowbase + k0 + 16);
  v16bf a;
#pragma unroll
  for (int i = 0; i < 8; ++i) { a[i] = lo[i]; a[i + 8] = hi[i]; }
  return a;
}

__device__ __forceinline__ v16bf load_b_frag(const __bf16* __restrict__ frag,
                                             int KT, int kt, int nt, int lane) {
  const __bf16* p = frag + ((size_t)((nt * KT + kt) * 32 + lane) << 4);
  v8bf lo = *(const v8bf*)p;
  v8bf hi = *(const v8bf*)(p + 8);
  v16bf b;
#pragma unroll
  for (int i = 0; i < 8; ++i) { b[i] = lo[i]; b[i + 8] = hi[i]; }
  return b;
}

__device__ __forceinline__ v8f vzero8() {
  v8f z;
#pragma unroll
  for (int i = 0; i < 8; ++i) z[i] = 0.0f;
  return z;
}

// ---------------------------------------------------------------------------
// Weight -> B-fragment pre-swizzle (f32 row-major [K][Nw] -> bf16 fragments)
// ---------------------------------------------------------------------------
__global__ __launch_bounds__(256) void prep_frag(const float* __restrict__ W,
                                                 __bf16* __restrict__ F,
                                                 int K, int Nw) {
  int t = blockIdx.x * blockDim.x + threadIdx.x;
  if (t >= K * Nw) return;
  int j = t & 15;
  int lane = (t >> 4) & 31;
  int tile = t >> 9;                 // 32*16 = 512 elements per (nt,kt) tile
  int KT = K >> 5;
  int kt = tile % KT;
  int nt = tile / KT;
  int krow = kt * 32 + ((lane & 16) ? 8 : 0) + (j < 8 ? j : j + 8);
  int ncol = nt * 16 + (lane & 15);
  F[t] = tobf(W[krow * Nw + ncol]);
}

// ---------------------------------------------------------------------------
// Gate path:  g = leaky_relu(X @ W1 + b1) @ w2 + b2        (rows x K -> rows)
// One wave per 16-row tile; 16 N-tiles * (K/32) WMMAs; 256->1 fused as a
// per-column dot + 16-lane butterfly reduce.  No LDS.
// ---------------------------------------------------------------------------
template <int K>
__global__ __launch_bounds__(128) void gate_kernel(
    const __bf16* __restrict__ X, const __bf16* __restrict__ W1f,
    const float* __restrict__ b1, const float* __restrict__ w2,
    const float* __restrict__ b2, float* __restrict__ gate) {
  constexpr int KT = K / 32;
  int lane = threadIdx.x & 31;
  int wave = threadIdx.x >> 5;
  int tile = blockIdx.x * 4 + wave;
  int row0 = tile * 16;
  const __bf16* rowp = X + (size_t)(row0 + (lane & 15)) * K;

  v16bf a[KT];
#pragma unroll
  for (int k = 0; k < KT; ++k) a[k] = load_a_frag(rowp, k, lane);

  float gp[8];
#pragma unroll
  for (int r = 0; r < 8; ++r) gp[r] = 0.0f;

  for (int nt = 0; nt < 16; ++nt) {
    v8f acc = vzero8();
#pragma unroll
    for (int k = 0; k < KT; ++k) {
      v16bf b = load_b_frag(W1f, KT, k, nt, lane);
      acc = __builtin_amdgcn_wmma_f32_16x16x32_bf16(false, a[k], false, b,
                                                    (short)0, acc, false, false);
    }
    int col = nt * 16 + (lane & 15);
    float bias = b1[col];
    float w2c = w2[col];
#pragma unroll
    for (int r = 0; r < 8; ++r) {
      float h = acc[r] + bias;
      h = (h < 0.0f) ? kSlope * h : h;
      gp[r] += h * w2c;
    }
  }
#pragma unroll
  for (int m = 1; m < 16; m <<= 1) {
#pragma unroll
    for (int r = 0; r < 8; ++r) gp[r] += __shfl_xor(gp[r], m, 16);
  }
  float bo = b2[0];
  if ((lane & 15) == 0) {
    int base = row0 + (lane ? 8 : 0);  // lane 0 -> rows 0..7, lane 16 -> 8..15
#pragma unroll
    for (int r = 0; r < 8; ++r) gate[base + r] = gp[r] + bo;
  }
}

// ---------------------------------------------------------------------------
// Softmax-attention weights per segment of 5 consecutive rows.
// ---------------------------------------------------------------------------
__global__ __launch_bounds__(256) void alpha_kernel(
    const float* __restrict__ gate, const float* __restrict__ weights,
    const int* __restrict__ widx, const float* __restrict__ powp,
    float* __restrict__ alpha, int nseg) {
  int s = blockIdx.x * blockDim.x + threadIdx.x;
  if (s >= nseg) return;
  float pw = powp[0];
  float g[kNpc], wf[kNpc];
  float gmax = -3.4e38f;
#pragma unroll
  for (int j = 0; j < kNpc; ++j) {
    int r = s * kNpc + j;
    g[j] = gate[r];
    gmax = fmaxf(gmax, g[j]);
    float w = weights[widx ? widx[r] : r];
    wf[j] = (pw > 0.0f) ? powf(w, pw) : 1.0f / (powf(w, fabsf(pw)) + 1e-10f);
  }
  float den = 0.0f;
#pragma unroll
  for (int j = 0; j < kNpc; ++j) {
    float e = wf[j] * expf(g[j] - gmax);
    g[j] = e;
    den += e;
  }
  float inv = 1.0f / (den + 1e-10f);
#pragma unroll
  for (int j = 0; j < kNpc; ++j) alpha[s * kNpc + j] = g[j] * inv;
}

// ---------------------------------------------------------------------------
// Msg path: m = leaky_relu(X @ W1 + b1) @ W2 + b2, then scatter
//   out[seg[row]] += alpha[row] * scale * m[row]   (atomic f32)
// h2 staged per-wave in LDS (16x256 bf16) between the two GEMMs.
// ---------------------------------------------------------------------------
template <int K>
__global__ __launch_bounds__(128) void msg_kernel(
    const __bf16* __restrict__ X, const __bf16* __restrict__ W1f,
    const float* __restrict__ b1, const __bf16* __restrict__ W2f,
    const float* __restrict__ b2, const float* __restrict__ alpha,
    const int* __restrict__ seg, float* __restrict__ out, float scale) {
  constexpr int KT = K / 32;
  __shared__ alignas(16) __bf16 hs[4][16 * kHid];
  int lane = threadIdx.x & 31;
  int wave = threadIdx.x >> 5;
  int tile = blockIdx.x * 4 + wave;
  int row0 = tile * 16;
  const __bf16* rowp = X + (size_t)(row0 + (lane & 15)) * K;
  __bf16* h = hs[wave];

  v16bf a[KT];
#pragma unroll
  for (int k = 0; k < KT; ++k) a[k] = load_a_frag(rowp, k, lane);

  for (int nt = 0; nt < 16; ++nt) {
    v8f acc = vzero8();
#pragma unroll
    for (int k = 0; k < KT; ++k) {
      v16bf b = load_b_frag(W1f, KT, k, nt, lane);
      acc = __builtin_amdgcn_wmma_f32_16x16x32_bf16(false, a[k], false, b,
                                                    (short)0, acc, false, false);
    }
    int col = nt * 16 + (lane & 15);
    float bias = b1[col];
#pragma unroll
    for (int r = 0; r < 8; ++r) {
      float v = acc[r] + bias;
      v = (v < 0.0f) ? kSlope * v : v;
      int rowi = r + ((lane & 16) ? 8 : 0);
      h[rowi * kHid + col] = tobf(v);
    }
  }
  __syncthreads();  // uniform control flow; orders DS writes before reads

  float alr[8];
  int sgo[8];
#pragma unroll
  for (int r = 0; r < 8; ++r) {
    int row = row0 + r + ((lane & 16) ? 8 : 0);
    alr[r] = alpha[row] * scale;
    sgo[r] = seg[row] * kFea;
  }

  const __bf16* hrow = h + (lane & 15) * kHid;
  for (int nt = 0; nt < 4; ++nt) {
    v8f acc = vzero8();
#pragma unroll
    for (int k = 0; k < 8; ++k) {            // K = 256
      v16bf a2 = load_a_frag(hrow, k, lane); // ds_load_b128 from LDS
      v16bf b = load_b_frag(W2f, 8, k, nt, lane);
      acc = __builtin_amdgcn_wmma_f32_16x16x32_bf16(false, a2, false, b,
                                                    (short)0, acc, false, false);
    }
    int col = nt * 16 + (lane & 15);
    float bias = b2[col];
#pragma unroll
    for (int r = 0; r < 8; ++r) {
      float v = (acc[r] + bias) * alr[r];
      atomicAdd(&out[sgo[r] + col], v);      // global_atomic_add_f32
    }
  }
}

// ---------------------------------------------------------------------------
// Elementwise kernels
// ---------------------------------------------------------------------------
__global__ __launch_bounds__(256) void embed_kernel(
    const float* __restrict__ ef, const float* __restrict__ ew,
    const float* __restrict__ W, const float* __restrict__ b,
    float* __restrict__ fea) {
  int t = blockIdx.x * blockDim.x + threadIdx.x;
  if (t >= kN * kFea) return;
  int n = t >> 6, j = t & 63;
  float v;
  if (j == 63) {
    v = ew[n];
  } else {
    float s = b[j];
    const float* x = ef + (size_t)n * kEmb;
    for (int k = 0; k < kEmb; ++k) s += x[k] * W[k * 63 + j];
    v = s;
  }
  fea[t] = v;
}

__global__ __launch_bounds__(256) void cat_kernel(
    const float* __restrict__ fea, const int* __restrict__ selfi,
    const int* __restrict__ nbri, __bf16* __restrict__ cat) {
  int t = blockIdx.x * blockDim.x + threadIdx.x;
  if (t >= kM * 128) return;
  int e = t >> 7, k = t & 127;
  int src = (k < 64) ? selfi[e] : nbri[e];
  cat[t] = tobf(fea[src * kFea + (k & 63)]);
}

__global__ __launch_bounds__(256) void f32_to_bf16(const float* __restrict__ in,
                                                   __bf16* __restrict__ out,
                                                   int n) {
  int t = blockIdx.x * blockDim.x + threadIdx.x;
  if (t < n) out[t] = tobf(in[t]);
}

}  // namespace

// ---------------------------------------------------------------------------
extern "C" void kernel_launch(void* const* d_in, const int* in_sizes, int n_in,
                              void* d_out, int out_size, void* d_ws,
                              size_t ws_size, hipStream_t stream) {
  (void)n_in; (void)out_size; (void)ws_size;
  const float* elem_w = (const float*)d_in[0];   // (N,1)
  const float* elem_f = (const float*)d_in[1];   // (N,200)
  const int* selfi = (const int*)d_in[2];        // (M,)
  const int* nbri  = (const int*)d_in[3];        // (M,)
  const int* cryi  = (const int*)d_in[4];        // (N,)

  // Param leaf order: JAX dict pytrees flatten with sorted keys ->
  // cry(27), embedding(2), graphs(81).  Fall back to insertion order
  // (embedding, graphs, cry) if leaf 5 looks like embedding.W (200*63).
  bool sorted_keys = (in_sizes[5] == 64 * 256);
  int embW, embB, gbase, cbase;
  if (sorted_keys) { cbase = 5; embW = 32; embB = 33; gbase = 34; }
  else             { embW = 5;  embB = 6;  gbase = 7;  cbase = 7 + 81; }
  // Pool layout (9 leaves): gate.hid.W, gate.hid.b, gate.out.W, gate.out.b,
  //                         msg.hid.W,  msg.hid.b,  msg.out.W,  msg.out.b, pow
  auto pf = [&](int i) -> const float* { return (const float*)d_in[i]; };

  // Workspace layout (all offsets 256B aligned; ~235 MB total)
  char* ws = (char*)d_ws;
  float* feaA = (float*)ws;                                 // N*64 f32
  float* feaB = (float*)(ws + 32768000);                    // N*64 f32
  __bf16* catb = (__bf16*)(ws + 65536000);                  // M*128 bf16
  float* gateb = (float*)(ws + 65536000 + 163840000);       // M f32
  float* alphab = gateb + kM;                               // M f32
  __bf16* fragG = (__bf16*)((char*)(alphab + kM));          // <= 128*256 bf16
  __bf16* fragM1 = fragG + 128 * 256;
  __bf16* fragM2 = fragM1 + 128 * 256;                      // 256*64 bf16

  const float third = 1.0f / 3.0f;

  // 1) Embedding: fea = [elem_fea @ Wemb + bemb, elem_weights]
  embed_kernel<<<(kN * kFea + 255) / 256, 256, 0, stream>>>(
      elem_f, elem_w, pf(embW), pf(embB), feaA);

  float* cur = feaA;
  float* nxt = feaB;

  // 2) Three message-passing layers, three heads each
  for (int g = 0; g < 3; ++g) {
    cat_kernel<<<(kM * 128 + 255) / 256, 256, 0, stream>>>(cur, selfi, nbri,
                                                           catb);
    hipMemcpyAsync(nxt, cur, (size_t)kN * kFea * sizeof(float),
                   hipMemcpyDeviceToDevice, stream);  // fea residual
    for (int h = 0; h < 3; ++h) {
      int pb = gbase + (g * 3 + h) * 9;
      prep_frag<<<(128 * 256 + 255) / 256, 256, 0, stream>>>(pf(pb + 0), fragG,
                                                             128, 256);
      prep_frag<<<(128 * 256 + 255) / 256, 256, 0, stream>>>(pf(pb + 4), fragM1,
                                                             128, 256);
      prep_frag<<<(256 * 64 + 255) / 256, 256, 0, stream>>>(pf(pb + 6), fragM2,
                                                            256, 64);
      gate_kernel<128><<<kM / 64, 128, 0, stream>>>(catb, fragG, pf(pb + 1),
                                                    pf(pb + 2), pf(pb + 3),
                                                    gateb);
      alpha_kernel<<<(kN + 255) / 256, 256, 0, stream>>>(gateb, elem_w, nbri,
                                                         pf(pb + 8), alphab,
                                                         kN);
      msg_kernel<128><<<kM / 64, 128, 0, stream>>>(catb, fragM1, pf(pb + 5),
                                                   fragM2, pf(pb + 7), alphab,
                                                   selfi, nxt, third);
    }
    float* tmp = cur; cur = nxt; nxt = tmp;
  }

  // 3) Crystal pooling heads -> d_out (C,64), mean of 3 heads
  f32_to_bf16<<<(kN * kFea + 255) / 256, 256, 0, stream>>>(cur, catb,
                                                           kN * kFea);
  hipMemsetAsync(d_out, 0, (size_t)kC * kFea * sizeof(float), stream);
  for (int h = 0; h < 3; ++h) {
    int pb = cbase + h * 9;
    prep_frag<<<(64 * 256 + 255) / 256, 256, 0, stream>>>(pf(pb + 0), fragG, 64,
                                                          256);
    prep_frag<<<(64 * 256 + 255) / 256, 256, 0, stream>>>(pf(pb + 4), fragM1,
                                                          64, 256);
    prep_frag<<<(256 * 64 + 255) / 256, 256, 0, stream>>>(pf(pb + 6), fragM2,
                                                          256, 64);
    gate_kernel<64><<<kN / 64, 128, 0, stream>>>(catb, fragG, pf(pb + 1),
                                                 pf(pb + 2), pf(pb + 3), gateb);
    alpha_kernel<<<(kC + 255) / 256, 256, 0, stream>>>(gateb, elem_w, nullptr,
                                                       pf(pb + 8), alphab, kC);
    msg_kernel<64><<<kN / 64, 128, 0, stream>>>(catb, fragM1, pf(pb + 5),
                                                fragM2, pf(pb + 7), alphab,
                                                cryi, (float*)d_out, third);
  }
}
